// DecodeMultiHeadAttention_54554674594303
// MI455X (gfx1250) — compile-verified, matrix operand order checked by reading, not run on hardware
//
#include <hip/hip_runtime.h>
#include <hip/hip_bf16.h>
#include <math.h>

typedef __attribute__((ext_vector_type(16))) _Float16 v16h;
typedef __attribute__((ext_vector_type(8)))  _Float16 v8h;
typedef __attribute__((ext_vector_type(8)))  float    v8f;

static constexpr int HN = 16;    // heads
static constexpr int DH = 64;    // head size
static constexpr int TT = 1024;  // sequence
static constexpr int CC = 1024;  // embed
static constexpr int BB = 2;     // batch

// ---------------- fragment helpers (ISA 7.12.2 layouts) ----------------
// A (16xK f16, row-major src [rows, ld]): lane m = lane&15, halves:
//   i in [0,8)  -> k = k0 + (lane>>4)*8 + i
//   i in [8,16) -> k = k0 + 16 + (lane>>4)*8 + (i-8)
// B uses the identical map with n = lane&15 when B is stored as [N, K] row-major.
__device__ __forceinline__ v16h load_frag_f16(const _Float16* __restrict__ base,
                                              int ld, int r0, int k0) {
  const int lane = threadIdx.x & 31;
  const int r    = r0 + (lane & 15);
  const int hi   = (lane >> 4) & 1;
  const _Float16* p = base + (size_t)r * ld + k0 + hi * 8;
  union { v16h v; v8h h[2]; } u;
  u.h[0] = *(const v8h*)(p);
  u.h[1] = *(const v8h*)(p + 16);
  return u.v;
}

__device__ __forceinline__ v16h load_frag_f32cvt(const float* __restrict__ base,
                                                 int ld, int r0, int k0) {
  const int lane = threadIdx.x & 31;
  const int r    = r0 + (lane & 15);
  const int hi   = (lane >> 4) & 1;
  const float* p = base + (size_t)r * ld + k0 + hi * 8;
  v16h v;
#pragma unroll
  for (int g = 0; g < 2; ++g)
#pragma unroll
    for (int j = 0; j < 8; ++j)
      v[g * 8 + j] = (_Float16)p[g * 16 + j];
  return v;
}

__device__ __forceinline__ v8f wmma16(v16h a, v16h b, v8f c) {
  return __builtin_amdgcn_wmma_f32_16x16x32_f16(false, a, false, b,
                                                (short)0, c, false, false);
}

// ---------------- stage 1: dtype conversion / weight transpose ----------------
__global__ __launch_bounds__(256) void cvt_f32_to_f16_kernel(
    const float* __restrict__ in, _Float16* __restrict__ out, int n) {
  for (int i = blockIdx.x * blockDim.x + threadIdx.x; i < n;
       i += gridDim.x * blockDim.x)
    out[i] = (_Float16)in[i];
}

// w: [H, C, D] fp32 -> wT: [H, D, C] f16
__global__ __launch_bounds__(256) void transpose_w_kernel(
    const float* __restrict__ w, _Float16* __restrict__ wT) {
  const int n = HN * CC * DH;
  for (int i = blockIdx.x * blockDim.x + threadIdx.x; i < n;
       i += gridDim.x * blockDim.x) {
    int h = i / (CC * DH);
    int r = i - h * (CC * DH);
    int c = r / DH;
    int d = r - c * DH;
    wT[((size_t)h * DH + d) * CC + c] = (_Float16)w[i];
  }
}

// ---------------- stage 2: QKV projection GEMMs ----------------
// xh: [B*T, C] f16 ; wT: [H, D, C] f16
// q,k -> [H, B, T, D] row-major ; v -> vT [H, B, D, T] (transposed for wei*V)
// Per K-step: issue all 5 fragment loads as one batch, then 4 WMMAs.
__global__ __launch_bounds__(256) void qkv_gemm_kernel(
    const _Float16* __restrict__ xh, const _Float16* __restrict__ wqT,
    const _Float16* __restrict__ wkT, const _Float16* __restrict__ wvT,
    _Float16* __restrict__ qh, _Float16* __restrict__ kh,
    _Float16* __restrict__ vT) {
  const int lane  = threadIdx.x & 31;
  const int wave  = threadIdx.x >> 5;
  const int which = blockIdx.z % 3;   // 0=q 1=k 2=v
  const int h     = blockIdx.z / 3;
  const int b     = blockIdx.y;
  const int t0    = (blockIdx.x * (blockDim.x >> 5) + wave) * 16;

  const _Float16* wT = (which == 0) ? wqT : (which == 1) ? wkT : wvT;
  const _Float16* wb = wT + (size_t)h * DH * CC;   // [D, C]
  const _Float16* xb = xh + (size_t)b * TT * CC;   // [T, C]

  v8f acc[4] = {};
  for (int k0 = 0; k0 < CC; k0 += 32) {
    // batch all loads for this K-step first (one VMEM clause)
    v16h a  = load_frag_f16(xb, CC, t0, k0);
    v16h b0 = load_frag_f16(wb, CC,  0, k0);
    v16h b1 = load_frag_f16(wb, CC, 16, k0);
    v16h b2 = load_frag_f16(wb, CC, 32, k0);
    v16h b3 = load_frag_f16(wb, CC, 48, k0);
    acc[0] = wmma16(a, b0, acc[0]);
    acc[1] = wmma16(a, b1, acc[1]);
    acc[2] = wmma16(a, b2, acc[2]);
    acc[3] = wmma16(a, b3, acc[3]);
  }

  const int n  = lane & 15;
  const int mb = ((lane >> 4) & 1) * 8;
  if (which < 2) {
    _Float16* ob = ((which == 0) ? qh : kh) + (size_t)(h * BB + b) * TT * DH;
#pragma unroll
    for (int j = 0; j < 4; ++j)
#pragma unroll
      for (int r = 0; r < 8; ++r)
        ob[(size_t)(t0 + mb + r) * DH + j * 16 + n] = (_Float16)acc[j][r];
  } else {
    _Float16* ob = vT + (size_t)(h * BB + b) * DH * TT;  // [D, T]
#pragma unroll
    for (int j = 0; j < 4; ++j)
#pragma unroll
      for (int r = 0; r < 8; ++r)
        ob[(size_t)(j * 16 + n) * TT + (t0 + mb + r)] = (_Float16)acc[j][r];
  }
}

// ---------------- stage 3: raw causal scores S = Q K^T * C^-0.5 ----------------
// NOTE: ALiBi term base**clip(j-i,0) == 1.0 on every unmasked (j<=i) entry,
// i.e. a constant pre-softmax shift -> softmax-invariant -> omitted.
__global__ __launch_bounds__(32) void scores_kernel(
    const _Float16* __restrict__ qh, const _Float16* __restrict__ kh,
    float* __restrict__ wei) {
  const int lane = threadIdx.x & 31;
  const int qt = blockIdx.x, b = blockIdx.y, h = blockIdx.z;
  const int t0 = qt * 16;
  const _Float16* qb = qh + (size_t)(h * BB + b) * TT * DH;
  const _Float16* kb = kh + (size_t)(h * BB + b) * TT * DH;
  float* wb = wei + (size_t)(h * BB + b) * TT * TT;
  const float scale = 0.03125f;  // 1024^-0.5 (reference scales by n_embd^-0.5)

  const v16h a0 = load_frag_f16(qb, DH, t0, 0);   // Q fragments reused over s
  const v16h a1 = load_frag_f16(qb, DH, t0, 32);
  const int n  = lane & 15;
  const int mb = ((lane >> 4) & 1) * 8;

  // double-buffered over key tiles (lower-triangular tiles only)
  v16h b0 = load_frag_f16(kb, DH, 0, 0);
  v16h b1 = load_frag_f16(kb, DH, 0, 32);
  for (int st = 0; st < qt; ++st) {
    const int s0 = st * 16;
    v16h n0f = load_frag_f16(kb, DH, s0 + 16, 0);
    v16h n1f = load_frag_f16(kb, DH, s0 + 16, 32);
    v8f acc = {};
    acc = wmma16(a0, b0, acc);
    acc = wmma16(a1, b1, acc);
#pragma unroll
    for (int r = 0; r < 8; ++r)
      wb[(size_t)(t0 + mb + r) * TT + s0 + n] = acc[r] * scale;
    b0 = n0f;
    b1 = n1f;
  }
  {  // diagonal tile
    const int s0 = qt * 16;
    v8f acc = {};
    acc = wmma16(a0, b0, acc);
    acc = wmma16(a1, b1, acc);
#pragma unroll
    for (int r = 0; r < 8; ++r)
      wb[(size_t)(t0 + mb + r) * TT + s0 + n] = acc[r] * scale;
  }
}

// ---------------- stage 4: row softmax over [0,i], zero-fill (i,T) ----------------
__global__ __launch_bounds__(256) void softmax_rows_kernel(
    float* __restrict__ wei) {
  const size_t row = blockIdx.x;          // over H*B*T rows
  float* p = wei + row * (size_t)TT;
  const int i = (int)(row % TT);
  const int len = i + 1;
  __shared__ float red[256];

  float m = -INFINITY;
  for (int j = threadIdx.x; j < len; j += blockDim.x) m = fmaxf(m, p[j]);
  red[threadIdx.x] = m;
  __syncthreads();
  for (int s = 128; s > 0; s >>= 1) {
    if ((int)threadIdx.x < s)
      red[threadIdx.x] = fmaxf(red[threadIdx.x], red[threadIdx.x + s]);
    __syncthreads();
  }
  m = red[0];
  __syncthreads();

  float sum = 0.f;
  for (int j = threadIdx.x; j < len; j += blockDim.x) {
    float e = __expf(p[j] - m);
    p[j] = e;
    sum += e;
  }
  red[threadIdx.x] = sum;
  __syncthreads();
  for (int s = 128; s > 0; s >>= 1) {
    if ((int)threadIdx.x < s) red[threadIdx.x] += red[threadIdx.x + s];
    __syncthreads();
  }
  const float inv = 1.f / red[0];
  for (int j = threadIdx.x; j < len; j += blockDim.x) p[j] *= inv;
  for (int j = len + threadIdx.x; j < TT; j += blockDim.x) p[j] = 0.f;  // masked
}

// ---------------- stage 5: out_heads = wei * V  (heads concatenated) ----------------
// wei rows are zero past the diagonal (stage 4), so full 32-wide K tiles are safe.
__global__ __launch_bounds__(32) void av_gemm_kernel(
    const float* __restrict__ wei, const _Float16* __restrict__ vT,
    _Float16* __restrict__ oh) {
  const int lane = threadIdx.x & 31;
  const int qt = blockIdx.x, b = blockIdx.y, h = blockIdx.z;
  const int t0 = qt * 16;
  const float* wb = wei + (size_t)(h * BB + b) * TT * TT;
  const _Float16* vb = vT + (size_t)(h * BB + b) * DH * TT;  // [D, T]

  v8f acc[4] = {};
  const int ksteps = qt / 2 + 1;  // cover K up to t0+16 (causal truncation)
  for (int ks = 0; ks < ksteps; ++ks) {
    const int k0 = ks * 32;
    v16h a  = load_frag_f32cvt(wb, TT, t0, k0);
    v16h b0 = load_frag_f16(vb, TT,  0, k0);
    v16h b1 = load_frag_f16(vb, TT, 16, k0);
    v16h b2 = load_frag_f16(vb, TT, 32, k0);
    v16h b3 = load_frag_f16(vb, TT, 48, k0);
    acc[0] = wmma16(a, b0, acc[0]);
    acc[1] = wmma16(a, b1, acc[1]);
    acc[2] = wmma16(a, b2, acc[2]);
    acc[3] = wmma16(a, b3, acc[3]);
  }

  _Float16* ob = oh + (size_t)b * TT * (HN * DH);
  const int n  = lane & 15;
  const int mb = ((lane >> 4) & 1) * 8;
#pragma unroll
  for (int j = 0; j < 4; ++j)
#pragma unroll
    for (int r = 0; r < 8; ++r)
      ob[(size_t)(t0 + mb + r) * (HN * DH) + h * DH + j * 16 + n] =
          (_Float16)acc[j][r];
}

// ---------------- stage 6: out = oh @ proj_w^T + proj_b ----------------
// proj_w is [out_f, in_f] row-major -> B fragment = rows of proj_w (K contiguous).
__global__ __launch_bounds__(32) void proj_gemm_kernel(
    const _Float16* __restrict__ oh, const float* __restrict__ pw,
    const float* __restrict__ pb, float* __restrict__ out) {
  const int lane = threadIdx.x & 31;
  const int m0 = blockIdx.x * 16;
  const int n0 = blockIdx.y * 64;

  v8f acc[4] = {};
  for (int k0 = 0; k0 < CC; k0 += 32) {
    v16h a  = load_frag_f16(oh, CC, m0, k0);
    v16h b0 = load_frag_f32cvt(pw, CC, n0 +  0, k0);
    v16h b1 = load_frag_f32cvt(pw, CC, n0 + 16, k0);
    v16h b2 = load_frag_f32cvt(pw, CC, n0 + 32, k0);
    v16h b3 = load_frag_f32cvt(pw, CC, n0 + 48, k0);
    acc[0] = wmma16(a, b0, acc[0]);
    acc[1] = wmma16(a, b1, acc[1]);
    acc[2] = wmma16(a, b2, acc[2]);
    acc[3] = wmma16(a, b3, acc[3]);
  }

  const int n  = lane & 15;
  const int mb = ((lane >> 4) & 1) * 8;
#pragma unroll
  for (int j = 0; j < 4; ++j) {
    const float bias = pb[n0 + j * 16 + n];
#pragma unroll
    for (int r = 0; r < 8; ++r)
      out[(size_t)(m0 + mb + r) * CC + n0 + j * 16 + n] = acc[j][r] + bias;
  }
}

// ---------------- host ----------------
extern "C" void kernel_launch(void* const* d_in, const int* in_sizes, int n_in,
                              void* d_out, int out_size, void* d_ws, size_t ws_size,
                              hipStream_t stream) {
  const float* x  = (const float*)d_in[0];
  const float* wk = (const float*)d_in[1];   // NOTE dict order: wk before wq
  const float* wq = (const float*)d_in[2];
  const float* wv = (const float*)d_in[3];
  const float* pw = (const float*)d_in[4];
  const float* pb = (const float*)d_in[5];

  float* wei = (float*)d_out;                                   // [H,B,T,T]
  float* out = (float*)d_out + (size_t)HN * BB * TT * TT;       // [B,T,C]

  // workspace carve-up (f16 units)
  _Float16* ws  = (_Float16*)d_ws;
  const size_t XH_N = (size_t)BB * TT * CC;   // 2,097,152
  const size_t WT_N = (size_t)HN * DH * CC;   // 1,048,576
  const size_t QK_N = (size_t)HN * BB * TT * DH;
  _Float16* xh  = ws;
  _Float16* wqT = xh  + XH_N;
  _Float16* wkT = wqT + WT_N;
  _Float16* wvT = wkT + WT_N;
  _Float16* qh  = wvT + WT_N;
  _Float16* kh  = qh  + QK_N;
  _Float16* vT  = kh  + QK_N;
  _Float16* oh  = vT  + QK_N;

  // 1) conversions
  cvt_f32_to_f16_kernel<<<4096, 256, 0, stream>>>(x, xh, (int)XH_N);
  transpose_w_kernel<<<4096, 256, 0, stream>>>(wq, wqT);
  transpose_w_kernel<<<4096, 256, 0, stream>>>(wk, wkT);
  transpose_w_kernel<<<4096, 256, 0, stream>>>(wv, wvT);

  // 2) QKV projections: 8 waves/block, each wave one 16x64 tile
  qkv_gemm_kernel<<<dim3(TT / (16 * 8), BB, HN * 3), 256, 0, stream>>>(
      xh, wqT, wkT, wvT, qh, kh, vT);

  // 3) raw causal scores
  scores_kernel<<<dim3(TT / 16, BB, HN), 32, 0, stream>>>(qh, kh, wei);

  // 4) softmax per row (one block per row; H*B*T rows)
  softmax_rows_kernel<<<(unsigned)(HN * BB * TT), 256, 0, stream>>>(wei);

  // 5) wei * V -> concatenated heads (f16)
  av_gemm_kernel<<<dim3(TT / 16, BB, HN), 32, 0, stream>>>(wei, vT, oh);

  // 6) output projection + bias
  proj_gemm_kernel<<<dim3((BB * TT) / 16, CC / 64), 32, 0, stream>>>(
      oh, pw, pb, out);
}
